// MultiHeadedAttention_16793322127677
// MI455X (gfx1250) — matrix operand order repaired and checked by reading, not compiled, not run on hardware
//
#include <hip/hip_runtime.h>
#include <math.h>

// ---------------------------------------------------------------------------
// MultiHeadedAttention for MI455X (gfx1250, wave32, WMMA).
//
// Shapes: B=2, S=2048, D=1024, H=16, dk=64.  ~69 GFLOP total, ~100MB touched
// (fits in 192MB L2).  All matmuls run on v_wmma_f32_16x16x32_f16 (f16 inputs,
// f32 accumulate); softmax is an online (flash) reduction held in fp32
// registers so the 2048x2048 score matrix never hits memory.  Attention
// blocks are 4 waves sharing K/V tiles through LDS (b128-coalesced staging).
// ---------------------------------------------------------------------------

typedef __attribute__((ext_vector_type(16))) _Float16 v16h;
typedef __attribute__((ext_vector_type(8)))  float    v8f;

#define D_MODEL 1024
#define SEQ     2048
#define BATCH   2
#define NHEAD   16
#define DKH     64
#define NROWS   (BATCH*SEQ)      // 4096 flattened (b,s) rows
#define QTPB    4                // q-tiles (waves) per attention block

static __device__ __forceinline__ v8f wmma_f16(v16h a, v16h b, v8f c) {
  // 8 args: (neg_a, A, neg_b, B, c_mod, C, reuse_a, reuse_b)
  return __builtin_amdgcn_wmma_f32_16x16x32_f16(false, a, false, b, (short)0, c,
                                                false, false);
}

// ---------------------------------------------------------------------------
// Fragment layout used here (wave32, contiguous k for A and B):
//  A (16x32 f16):  lane&15 = row M; lanes<16 hold K 0-15,
//                  lanes>=16 hold K 16-31 (kb = 0/16).
//  B (32x16 f16):  lane&15 = col N; lanes<16 hold K rows 0-15,
//                  lanes>=16 hold K rows 16-31 (kb = 0/16).
//  C/D (16x16 f32): VGPR r holds row (mBase+r), lane&15 = col N,
//                  mBase = 0 (lanes<16) or 8 (lanes>=16).
// ---------------------------------------------------------------------------

// ============================================================================
// Stage 1: Q/K/V = X @ W^T + b   (z selects which projection)
//   out[n,m] = sum_k X[n,k] * W[m,k] + bias[m]   -- both operands k-contiguous
//   Output f16 in [B,H,S,dk] layout for the attention stage.
// ============================================================================
__global__ __launch_bounds__(32)
void qkv_proj_kernel(const float* __restrict__ Xq, const float* __restrict__ Xk,
                     const float* __restrict__ Xv,
                     const float* __restrict__ Wq, const float* __restrict__ bq,
                     const float* __restrict__ Wk, const float* __restrict__ bk,
                     const float* __restrict__ Wv, const float* __restrict__ bv,
                     _Float16* __restrict__ Q16, _Float16* __restrict__ K16,
                     _Float16* __restrict__ V16)
{
  const int lane = threadIdx.x;
  const int tn   = blockIdx.x;            // 16-row tile of N=4096
  const int tm   = blockIdx.y;            // 64-col tile of M=1024
  const int z    = blockIdx.z;            // 0=Q 1=K 2=V
  const float* X    = (z == 0) ? Xq : (z == 1) ? Xk : Xv;
  const float* W    = (z == 0) ? Wq : (z == 1) ? Wk : Wv;
  const float* bias = (z == 0) ? bq : (z == 1) ? bk : bv;
  _Float16*    Out  = (z == 0) ? Q16 : (z == 1) ? K16 : V16;

  const int row   = lane & 15;
  const int koff  = (lane < 16) ? 0 : 8;
  const int kb    = (lane < 16) ? 0 : 16;
  const int mcol  = lane & 15;
  const int n0    = tn * 16;

  v8f acc[4];
  #pragma unroll
  for (int s = 0; s < 4; ++s) {
    float bb = bias[tm * 64 + s * 16 + mcol];   // column-constant bias
    #pragma unroll
    for (int r = 0; r < 8; ++r) acc[s][r] = bb;
  }

  const float* xrow = X + (size_t)(n0 + row) * D_MODEL;

  for (int kk = 0; kk < D_MODEL; kk += 32) {
    __builtin_prefetch(xrow + kk + 128, 0, 0);   // global_prefetch next chunk
    v16h a;
    #pragma unroll
    for (int i = 0; i < 8; ++i) {
      a[i]     = (_Float16)xrow[kk + kb + i];
      a[i + 8] = (_Float16)xrow[kk + kb + 8 + i];
    }
    #pragma unroll
    for (int s = 0; s < 4; ++s) {
      const float* wrow = W + (size_t)(tm * 64 + s * 16 + mcol) * D_MODEL + kk + kb;
      v16h bfr;
      #pragma unroll
      for (int i = 0; i < 16; ++i) bfr[i] = (_Float16)wrow[i];
      acc[s] = wmma_f16(a, bfr, acc[s]);
    }
  }

  const int mBase = (lane < 16) ? 0 : 8;
  #pragma unroll
  for (int s = 0; s < 4; ++s) {
    int m = tm * 64 + s * 16 + mcol;
    int h = m >> 6, d = m & 63;
    #pragma unroll
    for (int r = 0; r < 8; ++r) {
      int n = n0 + mBase + r;
      int b = n >> 11, sq = n & (SEQ - 1);
      Out[(((size_t)(b * NHEAD + h)) * SEQ + sq) * DKH + d] = (_Float16)acc[s][r];
    }
  }
}

// ============================================================================
// Stage 2: flash attention per (b,h).  Block = 4 waves = 64 query rows;
// each 32-key chunk of K and V (4KB each) is staged to LDS with coalesced
// b128 loads and shared by all 4 waves.  Online softmax in fp32 registers;
// P transposed C-layout -> A-layout through a per-wave LDS buffer.
// Context written f16 as [B,S,H*dk].
// ============================================================================
__global__ __launch_bounds__(32 * QTPB)
void attn_kernel(const _Float16* __restrict__ Q16, const _Float16* __restrict__ K16,
                 const _Float16* __restrict__ V16, _Float16* __restrict__ CTX)
{
  __shared__ _Float16 ldsK[32 * DKH];          // 32 keys x 64 dk  (4KB)
  __shared__ _Float16 ldsV[32 * DKH];          // 32 keys x 64 dk  (4KB)
  __shared__ float    pbuf[QTPB][16 * 32];     // per-wave P staging (4x2KB)

  const int tid  = threadIdx.x;
  const int lane = tid & 31;
  const int wid  = tid >> 5;
  const int qt = blockIdx.x * QTPB + wid;
  const int h = blockIdx.y, b = blockIdx.z;
  const size_t base = ((size_t)(b * NHEAD + h)) * SEQ * DKH;
  const _Float16* Qh = Q16 + base;
  const _Float16* Kh = K16 + base;
  const _Float16* Vh = V16 + base;

  const int row   = lane & 15;
  const int koff  = (lane < 16) ? 0 : 8;
  const int kb    = (lane < 16) ? 0 : 16;
  const int mBase = (lane < 16) ? 0 : 8;
  const int col   = lane & 15;

  // Q fragments (d = 0..63) live in registers for the whole key sweep.
  v16h aq0, aq1;
  const _Float16* qrow = Qh + (size_t)(qt * 16 + row) * DKH;
  #pragma unroll
  for (int i = 0; i < 8; ++i) {
    aq0[i]     = qrow[kb + i];
    aq0[i + 8] = qrow[kb + 8 + i];
    aq1[i]     = qrow[32 + kb + i];
    aq1[i + 8] = qrow[40 + kb + i];
  }

  float mrow[8], lrow[8];
  v8f zero = {};
  v8f acc[4];
  #pragma unroll
  for (int r = 0; r < 8; ++r) { mrow[r] = -1e30f; lrow[r] = 0.0f; }
  #pragma unroll
  for (int d = 0; d < 4; ++d) acc[d] = zero;

  for (int kt = 0; kt < SEQ; kt += 32) {
    // ---- cooperative staging of K/V tile: 128 threads x 2 x 16B each ----
    __syncthreads();                      // previous iteration's LDS reads done
    {
      const uint4* gK = (const uint4*)(Kh + (size_t)kt * DKH);  // 256 x 16B
      const uint4* gV = (const uint4*)(Vh + (size_t)kt * DKH);
      uint4* sK = (uint4*)ldsK;
      uint4* sV = (uint4*)ldsV;
      sK[tid]       = gK[tid];
      sK[tid + 128] = gK[tid + 128];
      sV[tid]       = gV[tid];
      sV[tid + 128] = gV[tid + 128];
    }
    __syncthreads();

    float s0[8], s1[8];
    // --- score tile j=0: keys [kt, kt+16) --- (contiguous 32B LDS reads)
    {
      const _Float16* kr = ldsK + (size_t)col * DKH;
      v16h b0 = *(const v16h*)(kr + kb);
      v16h b1 = *(const v16h*)(kr + 32 + kb);
      v8f c = {};
      c = wmma_f16(aq0, b0, c);
      c = wmma_f16(aq1, b1, c);
      #pragma unroll
      for (int r = 0; r < 8; ++r) s0[r] = c[r] * 0.125f;   // 1/sqrt(64)
    }
    // --- score tile j=1: keys [kt+16, kt+32) ---
    {
      const _Float16* kr = ldsK + (size_t)(16 + col) * DKH;
      v16h b0 = *(const v16h*)(kr + kb);
      v16h b1 = *(const v16h*)(kr + 32 + kb);
      v8f c = {};
      c = wmma_f16(aq0, b0, c);
      c = wmma_f16(aq1, b1, c);
      #pragma unroll
      for (int r = 0; r < 8; ++r) s1[r] = c[r] * 0.125f;
    }

    // Online softmax update.  Each row's 16 columns live in one 16-lane half,
    // so reduce with xor-shuffles of width 16.
    float alpha[8];
    #pragma unroll
    for (int r = 0; r < 8; ++r) {
      float v = fmaxf(s0[r], s1[r]);
      #pragma unroll
      for (int off = 1; off < 16; off <<= 1) v = fmaxf(v, __shfl_xor(v, off, 16));
      float mnew = fmaxf(mrow[r], v);
      alpha[r] = __expf(mrow[r] - mnew);
      mrow[r]  = mnew;
      s0[r] = __expf(s0[r] - mnew);
      s1[r] = __expf(s1[r] - mnew);
      float sum = s0[r] + s1[r];
      #pragma unroll
      for (int off = 1; off < 16; off <<= 1) sum += __shfl_xor(sum, off, 16);
      lrow[r] = lrow[r] * alpha[r] + sum;
    }
    #pragma unroll
    for (int d = 0; d < 4; ++d)
      #pragma unroll
      for (int r = 0; r < 8; ++r) acc[d][r] *= alpha[r];

    // Transpose P from C-layout to A-layout via per-wave LDS buffer.
    __syncthreads();
    #pragma unroll
    for (int r = 0; r < 8; ++r) {
      pbuf[wid][(mBase + r) * 32 + col]      = s0[r];
      pbuf[wid][(mBase + r) * 32 + 16 + col] = s1[r];
    }
    __syncthreads();
    v16h ap;
    #pragma unroll
    for (int i = 0; i < 8; ++i) {
      ap[i]     = (_Float16)pbuf[wid][row * 32 + kb + i];
      ap[i + 8] = (_Float16)pbuf[wid][row * 32 + kb + 8 + i];
    }

    // O += P (16x32) x V (32x64): 4 WMMAs, V fragments from LDS.
    #pragma unroll
    for (int d = 0; d < 4; ++d) {
      const _Float16* vcol = ldsV + (size_t)kb * DKH + d * 16 + col;
      v16h bvf;
      #pragma unroll
      for (int i = 0; i < 16; ++i) bvf[i] = vcol[(size_t)i * DKH];
      acc[d] = wmma_f16(ap, bvf, acc[d]);
    }
  }

  // Normalize and store context as f16 in [B,S,H*dk].
  float inv[8];
  #pragma unroll
  for (int r = 0; r < 8; ++r) inv[r] = 1.0f / lrow[r];
  #pragma unroll
  for (int d = 0; d < 4; ++d) {
    #pragma unroll
    for (int r = 0; r < 8; ++r) {
      int q = qt * 16 + mBase + r;
      CTX[((size_t)(b * SEQ + q)) * D_MODEL + h * DKH + d * 16 + col] =
          (_Float16)(acc[d][r] * inv[r]);
    }
  }
}

// ============================================================================
// Stage 3: out = CTX @ Wo^T + bo  (f32 result to d_out)
// ============================================================================
__global__ __launch_bounds__(32)
void out_proj_kernel(const _Float16* __restrict__ CTX, const float* __restrict__ Wo,
                     const float* __restrict__ bo, float* __restrict__ Out)
{
  const int lane = threadIdx.x;
  const int tn = blockIdx.x, tm = blockIdx.y;
  const int row  = lane & 15;
  const int koff = (lane < 16) ? 0 : 8;
  const int kb   = (lane < 16) ? 0 : 16;
  const int mcol = lane & 15;

  v8f acc[4];
  #pragma unroll
  for (int s = 0; s < 4; ++s) {
    float bb = bo[tm * 64 + s * 16 + mcol];
    #pragma unroll
    for (int r = 0; r < 8; ++r) acc[s][r] = bb;
  }

  const _Float16* arow = CTX + (size_t)(tn * 16 + row) * D_MODEL;
  for (int kk = 0; kk < D_MODEL; kk += 32) {
    __builtin_prefetch(arow + kk + 128, 0, 0);
    v16h a;
    #pragma unroll
    for (int i = 0; i < 8; ++i) {
      a[i]     = arow[kk + kb + i];
      a[i + 8] = arow[kk + kb + 8 + i];
    }
    #pragma unroll
    for (int s = 0; s < 4; ++s) {
      const float* wrow = Wo + (size_t)(tm * 64 + s * 16 + mcol) * D_MODEL + kk + kb;
      v16h bfr;
      #pragma unroll
      for (int i = 0; i < 16; ++i) bfr[i] = (_Float16)wrow[i];
      acc[s] = wmma_f16(a, bfr, acc[s]);
    }
  }

  const int mBase = (lane < 16) ? 0 : 8;
  #pragma unroll
  for (int s = 0; s < 4; ++s) {
    int m = tm * 64 + s * 16 + mcol;
    #pragma unroll
    for (int r = 0; r < 8; ++r) {
      int n = tn * 16 + mBase + r;
      Out[(size_t)n * D_MODEL + m] = acc[s][r];
    }
  }
}

// ============================================================================
// Host launch.  Workspace: Q16/K16/V16 ([B,H,S,dk] f16, 8MB each) + CTX16
// ([B,S,D] f16, 8MB) = 32MB total.
// ============================================================================
extern "C" void kernel_launch(void* const* d_in, const int* in_sizes, int n_in,
                              void* d_out, int out_size, void* d_ws, size_t ws_size,
                              hipStream_t stream) {
  const float* q  = (const float*)d_in[0];
  const float* k  = (const float*)d_in[1];
  const float* v  = (const float*)d_in[2];
  const float* Wq = (const float*)d_in[3];
  const float* bq = (const float*)d_in[4];
  const float* Wk = (const float*)d_in[5];
  const float* bk = (const float*)d_in[6];
  const float* Wv = (const float*)d_in[7];
  const float* bv = (const float*)d_in[8];
  const float* Wo = (const float*)d_in[9];
  const float* bo = (const float*)d_in[10];
  float* out = (float*)d_out;

  const size_t NT = (size_t)NROWS * D_MODEL;
  _Float16* Q16 = (_Float16*)d_ws;
  _Float16* K16 = Q16 + NT;
  _Float16* V16 = K16 + NT;
  _Float16* CTX = V16 + NT;

  qkv_proj_kernel<<<dim3(NROWS / 16, D_MODEL / 64, 3), 32, 0, stream>>>(
      q, k, v, Wq, bq, Wk, bk, Wv, bv, Q16, K16, V16);
  attn_kernel<<<dim3(SEQ / (16 * QTPB), NHEAD, BATCH), 32 * QTPB, 0, stream>>>(
      Q16, K16, V16, CTX);
  out_proj_kernel<<<dim3(NROWS / 16, D_MODEL / 64, 1), 32, 0, stream>>>(CTX, Wo, bo, out);
}
